// Seq2SeqAttention_40123584479799
// MI455X (gfx1250) — matrix operand order repaired and checked by reading, not compiled, not run on hardware
//
#include <hip/hip_runtime.h>
#include <math.h>

// Model dims (fixed by the reference)
#define BB 32
#define SS 128
#define TT 64
#define DE 512
#define HE 512
#define EE 1024
#define HD 1024
#define VV 32000
#define NHD 8
#define HDIM 128

typedef __attribute__((ext_vector_type(16))) __bf16 v16bf;
typedef __attribute__((ext_vector_type(8)))  float  v8f;

__device__ __forceinline__ unsigned short f2bf(float f) {
  unsigned int u = __builtin_bit_cast(unsigned int, f);
  u += 0x7FFFu + ((u >> 16) & 1u);               // round-to-nearest-even
  return (unsigned short)(u >> 16);
}

union FragU { uint4 q[2]; v16bf v; };

// ---------------------------------------------------------------------------
// WMMA GEMM: C[M,N] = A[M,K] * W[N,K]^T (+bias[N]) (+tanh)
// A, W row-major bf16 (ushort bits). Requires M%32==0, N%256==0, K%32==0.
// grid = (N/256, M/32), block = 128 (4 waves). Each wave: 32x64 tile.
// ---------------------------------------------------------------------------
__global__ __launch_bounds__(128) void k_gemm(
    const unsigned short* __restrict__ A, int lda,
    const unsigned short* __restrict__ W, int ldb,
    float* __restrict__ Cf, unsigned short* __restrict__ Cb, int ldc,
    const float* __restrict__ bias, int K, int act)
{
  const int lane = threadIdx.x & 31;
  const int wave = threadIdx.x >> 5;
  const int lr = lane & 15;     // row within 16 (A) / col within 16 (B)
  const int lh = lane >> 4;     // lane-half selects K sub-chunk
  const int rowBase = blockIdx.y * 32;
  const int colBase = blockIdx.x * 256 + wave * 64;

  const unsigned short* a0p = A + (size_t)(rowBase + lr) * lda;
  const unsigned short* a1p = A + (size_t)(rowBase + 16 + lr) * lda;
  const unsigned short* b0p = W + (size_t)(colBase +  0 + lr) * ldb;
  const unsigned short* b1p = W + (size_t)(colBase + 16 + lr) * ldb;
  const unsigned short* b2p = W + (size_t)(colBase + 32 + lr) * ldb;
  const unsigned short* b3p = W + (size_t)(colBase + 48 + lr) * ldb;

  v8f acc00 = {}, acc01 = {}, acc02 = {}, acc03 = {};
  v8f acc10 = {}, acc11 = {}, acc12 = {}, acc13 = {};

  for (int k0 = 0; k0 < K; k0 += 32) {
    FragU a0, a1, b0, b1, b2, b3;
    // A 16x32 fragment: lane half h holds contiguous K = {16h..16h+15}
    a0.q[0] = *(const uint4*)(a0p + k0 + lh * 16);
    a0.q[1] = *(const uint4*)(a0p + k0 + lh * 16 + 8);
    a1.q[0] = *(const uint4*)(a1p + k0 + lh * 16);
    a1.q[1] = *(const uint4*)(a1p + k0 + lh * 16 + 8);
    // B 32x16 fragment: lane half h holds contiguous K = {16h..16h+15}
    b0.q[0] = *(const uint4*)(b0p + k0 + lh * 16);
    b0.q[1] = *(const uint4*)(b0p + k0 + lh * 16 + 8);
    b1.q[0] = *(const uint4*)(b1p + k0 + lh * 16);
    b1.q[1] = *(const uint4*)(b1p + k0 + lh * 16 + 8);
    b2.q[0] = *(const uint4*)(b2p + k0 + lh * 16);
    b2.q[1] = *(const uint4*)(b2p + k0 + lh * 16 + 8);
    b3.q[0] = *(const uint4*)(b3p + k0 + lh * 16);
    b3.q[1] = *(const uint4*)(b3p + k0 + lh * 16 + 8);

    acc00 = __builtin_amdgcn_wmma_f32_16x16x32_bf16(false, a0.v, false, b0.v, (short)0, acc00, false, false);
    acc01 = __builtin_amdgcn_wmma_f32_16x16x32_bf16(false, a0.v, false, b1.v, (short)0, acc01, false, false);
    acc02 = __builtin_amdgcn_wmma_f32_16x16x32_bf16(false, a0.v, false, b2.v, (short)0, acc02, false, false);
    acc03 = __builtin_amdgcn_wmma_f32_16x16x32_bf16(false, a0.v, false, b3.v, (short)0, acc03, false, false);
    acc10 = __builtin_amdgcn_wmma_f32_16x16x32_bf16(false, a1.v, false, b0.v, (short)0, acc10, false, false);
    acc11 = __builtin_amdgcn_wmma_f32_16x16x32_bf16(false, a1.v, false, b1.v, (short)0, acc11, false, false);
    acc12 = __builtin_amdgcn_wmma_f32_16x16x32_bf16(false, a1.v, false, b2.v, (short)0, acc12, false, false);
    acc13 = __builtin_amdgcn_wmma_f32_16x16x32_bf16(false, a1.v, false, b3.v, (short)0, acc13, false, false);
  }

  // C/D layout: lane holds col = lr; rows = lh*8 + i
  auto store_tile = [&](v8f acc, int r0, int c0) {
    int col = c0 + lr;
    float bv = bias ? bias[col] : 0.0f;
#pragma unroll
    for (int i = 0; i < 8; ++i) {
      int row = r0 + lh * 8 + i;
      float v = acc[i] + bv;
      if (act == 1) v = tanhf(v);
      if (Cf) Cf[(size_t)row * ldc + col] = v;
      if (Cb) Cb[(size_t)row * ldc + col] = f2bf(v);
    }
  };
  store_tile(acc00, rowBase,      colBase);
  store_tile(acc01, rowBase,      colBase + 16);
  store_tile(acc02, rowBase,      colBase + 32);
  store_tile(acc03, rowBase,      colBase + 48);
  store_tile(acc10, rowBase + 16, colBase);
  store_tile(acc11, rowBase + 16, colBase + 16);
  store_tile(acc12, rowBase + 16, colBase + 32);
  store_tile(acc13, rowBase + 16, colBase + 48);
}

// ---------------------------------------------------------------------------
// Small elementwise kernels
// ---------------------------------------------------------------------------
__global__ void k_cvt(const float* __restrict__ in, unsigned short* __restrict__ out, int n) {
  int i = blockIdx.x * blockDim.x + threadIdx.x;
  if (i < n) out[i] = f2bf(in[i]);
}

__global__ void k_concat2(const float* __restrict__ A, int k1,
                          const float* __restrict__ B, int k2,
                          unsigned short* __restrict__ out, int total) {
  int i = blockIdx.x * blockDim.x + threadIdx.x;
  if (i >= total) return;
  int kk = k1 + k2;
  int r = i / kk, c = i - r * kk;
  float v = (c < k1) ? A[(size_t)r * k1 + c] : B[(size_t)r * k2 + (c - k1)];
  out[i] = f2bf(v);
}

__global__ void k_vadd(const float* __restrict__ a, const float* __restrict__ b,
                       float* __restrict__ o, int n) {
  int i = blockIdx.x * blockDim.x + threadIdx.x;
  if (i < n) o[i] = a[i] + b[i];
}

__global__ void k_zero(float* __restrict__ p, int n) {
  int i = blockIdx.x * blockDim.x + threadIdx.x;
  if (i < n) p[i] = 0.0f;
}

__global__ void k_gather(const int* __restrict__ tok, const float* __restrict__ emb,
                         unsigned short* __restrict__ out, int total, int dim) {
  int i = blockIdx.x * blockDim.x + threadIdx.x;
  if (i >= total) return;
  int r = i / dim, c = i - r * dim;
  out[i] = f2bf(emb[(size_t)tok[r] * dim + c]);
}

__global__ void k_copy_bf16(const unsigned short* __restrict__ src, int ss,
                            unsigned short* __restrict__ dst, int ds,
                            int rows, int cols) {
  int i = blockIdx.x * blockDim.x + threadIdx.x;
  if (i >= rows * cols) return;
  int r = i / cols, c = i - r * cols;
  dst[(size_t)r * ds + c] = src[(size_t)r * ss + c];
}

__device__ __forceinline__ float sigm(float x) { return 1.0f / (1.0f + __expf(-x)); }

// GRU gate math for one timestep. px pointer pre-offset to [b=0][t][0].
__global__ void k_gru_cell(const float* __restrict__ px, int pxs,
                           const float* __restrict__ gh,
                           float* __restrict__ h,
                           unsigned short* __restrict__ hbf, int hbs,
                           unsigned short* __restrict__ obf, int obs) {
  int i = blockIdx.x * blockDim.x + threadIdx.x;   // 32*512
  if (i >= BB * HE) return;
  int b = i >> 9, j = i & 511;
  const float* pr = px + (size_t)b * pxs;
  const float* gr = gh + (size_t)b * (3 * HE);
  float r = sigm(pr[j] + gr[j]);
  float z = sigm(pr[j + HE] + gr[j + HE]);
  float n = tanhf(pr[j + 2 * HE] + r * gr[j + 2 * HE]);
  float hv = (1.0f - z) * n + z * h[i];
  h[i] = hv;
  unsigned short hb = f2bf(hv);
  hbf[(size_t)b * hbs + j] = hb;
  obf[(size_t)b * obs + j] = hb;
}

// LSTM cell: g holds 4 gates (i,f,g,o) of width 1024 each; writes h (bf16) twice.
__global__ void k_lstm_cell(const float* __restrict__ g, float* __restrict__ c,
                            unsigned short* __restrict__ d0, int s0,
                            unsigned short* __restrict__ d1, int s1) {
  int i = blockIdx.x * blockDim.x + threadIdx.x;   // 32*1024
  if (i >= BB * HD) return;
  int b = i >> 10, j = i & 1023;
  const float* gr = g + (size_t)b * (4 * HD);
  float ig = sigm(gr[j]);
  float fg = sigm(gr[j + HD]);
  float gg = tanhf(gr[j + 2 * HD]);
  float og = sigm(gr[j + 3 * HD]);
  float c2 = fg * c[i] + ig * gg;
  float h2 = og * tanhf(c2);
  c[i] = c2;
  unsigned short hb = f2bf(h2);
  d0[(size_t)b * s0 + j] = hb;
  d1[(size_t)b * s1 + j] = hb;
}

// 8-head attention for one decoder step. Block per (b, head), thread = s / dim.
__global__ __launch_bounds__(128) void k_attention(const float* __restrict__ q,
                                                   const float* __restrict__ KV,
                                                   unsigned short* __restrict__ ctx) {
  __shared__ float sc[128];
  __shared__ float red[128];
  int b = blockIdx.x >> 3;
  int h = blockIdx.x & 7;
  int t = threadIdx.x;
  const float* qp = q + b * EE + h * HDIM;
  const float* kp = KV + (size_t)(b * SS + t) * (2 * EE) + h * HDIM;
  float acc = 0.0f;
#pragma unroll 4
  for (int d = 0; d < HDIM; ++d) acc = fmaf(qp[d], kp[d], acc);
  acc *= 0.088388347648318447f;   // 1/sqrt(128)
  red[t] = acc;
  __syncthreads();
  for (int off = 64; off > 0; off >>= 1) {
    if (t < off) red[t] = fmaxf(red[t], red[t + off]);
    __syncthreads();
  }
  float m = red[0];
  __syncthreads();
  float e = __expf(acc - m);
  sc[t] = e;
  red[t] = e;
  __syncthreads();
  for (int off = 64; off > 0; off >>= 1) {
    if (t < off) red[t] += red[t + off];
    __syncthreads();
  }
  float inv = 1.0f / red[0];
  const float* vp = KV + (size_t)(b * SS) * (2 * EE) + EE + h * HDIM + t;
  float cacc = 0.0f;
  for (int s = 0; s < SS; ++s) cacc = fmaf(sc[s], vp[(size_t)s * (2 * EE)], cacc);
  ctx[b * EE + h * HDIM + t] = f2bf(cacc * inv);
}

// ---------------------------------------------------------------------------
// Host side
// ---------------------------------------------------------------------------
static inline int nb(int n) { return (n + 255) / 256; }

static void gemm(hipStream_t s, const void* A, int lda, const void* W, int ldb,
                 float* Cf, void* Cb, int ldc, const float* bias,
                 int M, int N, int K, int act) {
  dim3 g((unsigned)(N / 256), (unsigned)(M / 32)), blk(128);
  k_gemm<<<g, blk, 0, s>>>((const unsigned short*)A, lda, (const unsigned short*)W,
                           ldb, Cf, (unsigned short*)Cb, ldc, bias, K, act);
}

extern "C" void kernel_launch(void* const* d_in, const int* in_sizes, int n_in,
                              void* d_out, int out_size, void* d_ws, size_t ws_size,
                              hipStream_t stream) {
  (void)in_sizes; (void)n_in; (void)out_size; (void)ws_size;
  const int*   src       = (const int*)d_in[0];
  const int*   tgt       = (const int*)d_in[1];
  const float* enc_emb   = (const float*)d_in[2];
  const float* g0Wih     = (const float*)d_in[3];
  const float* g0Whh     = (const float*)d_in[4];
  const float* g0bih     = (const float*)d_in[5];
  const float* g0bhh     = (const float*)d_in[6];
  const float* g1Wih     = (const float*)d_in[7];
  const float* g1Whh     = (const float*)d_in[8];
  const float* g1bih     = (const float*)d_in[9];
  const float* g1bhh     = (const float*)d_in[10];
  const float* dec_emb   = (const float*)d_in[11];
  const float* proj_W    = (const float*)d_in[12];
  const float* proj_b    = (const float*)d_in[13];
  const float* attn_in_W = (const float*)d_in[14];
  const float* attn_in_b = (const float*)d_in[15];
  const float* attn_o_W  = (const float*)d_in[16];
  const float* attn_o_b  = (const float*)d_in[17];
  const float* l0Wih     = (const float*)d_in[18];
  const float* l0Whh     = (const float*)d_in[19];
  const float* l0bih     = (const float*)d_in[20];
  const float* l0bhh     = (const float*)d_in[21];
  const float* l1Wih     = (const float*)d_in[22];
  const float* l1Whh     = (const float*)d_in[23];
  const float* l1bih     = (const float*)d_in[24];
  const float* l1bhh     = (const float*)d_in[25];
  const float* fc_W      = (const float*)d_in[26];
  const float* fc_b      = (const float*)d_in[27];
  float* out = (float*)d_out;

  // --- workspace bump allocator (offsets identical every call) ---
  char* p = (char*)d_ws;
  auto alloc = [&](size_t bytes) -> void* {
    void* r = (void*)p;
    p += (bytes + 255) & ~(size_t)255;
    return r;
  };
  typedef unsigned short us;
  us* w_g0Wih = (us*)alloc((size_t)2 * 3 * HE * DE * 2);
  us* w_g0Whh = (us*)alloc((size_t)2 * 3 * HE * HE * 2);
  us* w_g1Wih = (us*)alloc((size_t)2 * 3 * HE * (2 * HE) * 2);
  us* w_g1Whh = (us*)alloc((size_t)2 * 3 * HE * HE * 2);
  us* w_proj  = (us*)alloc((size_t)HD * EE * 2);
  us* w_ain   = (us*)alloc((size_t)3 * EE * EE * 2);
  us* w_aout  = (us*)alloc((size_t)EE * EE * 2);
  us* w_l0    = (us*)alloc((size_t)4 * HD * (DE + EE + HD) * 2);   // 4096 x 2560
  us* w_l1    = (us*)alloc((size_t)4 * HD * (2 * HD) * 2);         // 4096 x 2048
  us* w_fc    = (us*)alloc((size_t)VV * HD * 2);
  float* b_l0 = (float*)alloc((size_t)4 * HD * 4);
  float* b_l1 = (float*)alloc((size_t)4 * HD * 4);
  us* enc_x   = (us*)alloc((size_t)BB * SS * DE * 2);
  us* dec_x   = (us*)alloc((size_t)BB * TT * DE * 2);
  us* o0      = (us*)alloc((size_t)BB * SS * EE * 2);
  us* enc_o   = (us*)alloc((size_t)BB * SS * EE * 2);
  us* encHid  = (us*)alloc((size_t)64 * EE * 2);
  float* px   = (float*)alloc((size_t)BB * SS * 3 * HE * 4);
  float* gh   = (float*)alloc((size_t)BB * 3 * HE * 4);
  float* h32  = (float*)alloc((size_t)BB * HE * 4);
  float* KV   = (float*)alloc((size_t)BB * SS * 2 * EE * 4);
  float* qb   = (float*)alloc((size_t)BB * EE * 4);
  us* hinitb  = (us*)alloc((size_t)64 * HD * 2);
  us* xinA    = (us*)alloc((size_t)BB * (DE + EE + HD) * 2);       // 32 x 2560
  us* lstm1A  = (us*)alloc((size_t)BB * 2 * HD * 2);               // 32 x 2048
  us* ctxb    = (us*)alloc((size_t)BB * EE * 2);
  float* g4   = (float*)alloc((size_t)BB * 4 * HD * 4);
  float* c0   = (float*)alloc((size_t)BB * HD * 4);
  float* c1   = (float*)alloc((size_t)BB * HD * 4);
  us* dec_o   = (us*)alloc((size_t)BB * TT * HD * 2);

  // --- weight conversions (fp32 -> bf16) ---
  k_cvt<<<nb(2*3*HE*DE), 256, 0, stream>>>(g0Wih, w_g0Wih, 2*3*HE*DE);
  k_cvt<<<nb(2*3*HE*HE), 256, 0, stream>>>(g0Whh, w_g0Whh, 2*3*HE*HE);
  k_cvt<<<nb(2*3*HE*2*HE), 256, 0, stream>>>(g1Wih, w_g1Wih, 2*3*HE*2*HE);
  k_cvt<<<nb(2*3*HE*HE), 256, 0, stream>>>(g1Whh, w_g1Whh, 2*3*HE*HE);
  k_cvt<<<nb(HD*EE), 256, 0, stream>>>(proj_W, w_proj, HD*EE);
  k_cvt<<<nb(3*EE*EE), 256, 0, stream>>>(attn_in_W, w_ain, 3*EE*EE);
  k_cvt<<<nb(EE*EE), 256, 0, stream>>>(attn_o_W, w_aout, EE*EE);
  k_cvt<<<nb(VV*HD), 256, 0, stream>>>(fc_W, w_fc, VV*HD);
  k_concat2<<<nb(4*HD*(DE+EE+HD)), 256, 0, stream>>>(l0Wih, DE+EE, l0Whh, HD, w_l0, 4*HD*(DE+EE+HD));
  k_concat2<<<nb(4*HD*2*HD), 256, 0, stream>>>(l1Wih, HD, l1Whh, HD, w_l1, 4*HD*2*HD);
  k_vadd<<<nb(4*HD), 256, 0, stream>>>(l0bih, l0bhh, b_l0, 4*HD);
  k_vadd<<<nb(4*HD), 256, 0, stream>>>(l1bih, l1bhh, b_l1, 4*HD);

  // --- embeddings + state init ---
  k_gather<<<nb(BB*SS*DE), 256, 0, stream>>>(src, enc_emb, enc_x, BB*SS*DE, DE);
  k_gather<<<nb(BB*TT*DE), 256, 0, stream>>>(tgt, dec_emb, dec_x, BB*TT*DE, DE);
  k_zero<<<nb(64*EE/2), 256, 0, stream>>>((float*)encHid, 64*EE/2);  // bf16 zeros
  k_zero<<<nb(BB*HD), 256, 0, stream>>>(c0, BB*HD);
  k_zero<<<nb(BB*HD), 256, 0, stream>>>(c1, BB*HD);

  // --- Encoder: 2-layer bidirectional GRU ---
  for (int L = 0; L < 2; ++L) {
    const us* inA = (L == 0) ? enc_x : o0;
    int Kin = (L == 0) ? DE : EE;
    const us* Wih = (L == 0) ? w_g0Wih : w_g1Wih;
    const us* Whh = (L == 0) ? w_g0Whh : w_g1Whh;
    const float* bih = (L == 0) ? g0bih : g1bih;
    const float* bhh = (L == 0) ? g0bhh : g1bhh;
    us* obuf = (L == 0) ? o0 : enc_o;
    for (int d = 0; d < 2; ++d) {
      // input projection for all timesteps: (4096 x 3H)
      gemm(stream, inA, Kin, Wih + (size_t)d * 3 * HE * Kin, Kin,
           px, nullptr, 3 * HE, bih + d * 3 * HE, BB * SS, 3 * HE, Kin, 0);
      k_zero<<<nb(BB*HE), 256, 0, stream>>>(h32, BB * HE);
      us* hbf = encHid + (size_t)L * 32 * EE + d * HE;   // strided (lda=1024) state
      for (int i = 0; i < SS; ++i) {
        int t = d ? (SS - 1 - i) : i;
        gemm(stream, hbf, EE, Whh + (size_t)d * 3 * HE * HE, HE,
             gh, nullptr, 3 * HE, bhh + d * 3 * HE, BB, 3 * HE, HE, 0);
        k_gru_cell<<<nb(BB*HE), 256, 0, stream>>>(
            px + (size_t)t * 3 * HE, SS * 3 * HE, gh, h32,
            hbf, EE, obuf + (size_t)t * EE + d * HE, SS * EE);
      }
    }
  }

  // --- Decoder init: hinit = tanh(enc_hidden @ proj_W.T + proj_b) ---
  gemm(stream, encHid, EE, w_proj, EE, nullptr, hinitb, HD, proj_b, 64, HD, EE, 1);
  // scatter h0 -> xinA[:,1536:2560] and lstm1A[:,0:1024]; h1 -> lstm1A[:,1024:2048]
  k_copy_bf16<<<nb(BB*HD), 256, 0, stream>>>(hinitb, HD, xinA + DE + EE, DE + EE + HD, BB, HD);
  k_copy_bf16<<<nb(BB*HD), 256, 0, stream>>>(hinitb, HD, lstm1A, 2 * HD, BB, HD);
  k_copy_bf16<<<nb(BB*HD), 256, 0, stream>>>(hinitb + (size_t)32 * HD, HD, lstm1A + HD, 2 * HD, BB, HD);

  // --- Hoisted K/V projections: [Kh|Vh] = enc_out @ [Wk;Wv].T + [bk;bv] ---
  gemm(stream, enc_o, EE, w_ain + (size_t)EE * EE, EE, KV, nullptr, 2 * EE,
       attn_in_b + EE, BB * SS, 2 * EE, EE, 0);

  // --- Decoder: 64 steps ---
  for (int t = 0; t < TT; ++t) {
    // q = h1 @ Wq.T + bq   (scale folded into attention kernel)
    gemm(stream, lstm1A + HD, 2 * HD, w_ain, EE, qb, nullptr, EE, attn_in_b, BB, EE, EE, 0);
    k_attention<<<BB * NHD, 128, 0, stream>>>(qb, KV, ctxb);
    // attn_out -> xinA[:,512:1536]
    gemm(stream, ctxb, EE, w_aout, EE, nullptr, xinA + DE, DE + EE + HD,
         attn_o_b, BB, EE, EE, 0);
    // et -> xinA[:,0:512]
    k_copy_bf16<<<nb(BB*DE), 256, 0, stream>>>(dec_x + (size_t)t * DE, TT * DE, xinA, DE + EE + HD, BB, DE);
    // LSTM0: g = [xin, h0] @ [Wih|Whh].T + (bih+bhh)
    gemm(stream, xinA, DE + EE + HD, w_l0, DE + EE + HD, g4, nullptr, 4 * HD,
         b_l0, BB, 4 * HD, DE + EE + HD, 0);
    k_lstm_cell<<<nb(BB*HD), 256, 0, stream>>>(g4, c0, xinA + DE + EE, DE + EE + HD, lstm1A, 2 * HD);
    // LSTM1: g = [h0, h1] @ [Wih|Whh].T + (bih+bhh)
    gemm(stream, lstm1A, 2 * HD, w_l1, 2 * HD, g4, nullptr, 4 * HD,
         b_l1, BB, 4 * HD, 2 * HD, 0);
    k_lstm_cell<<<nb(BB*HD), 256, 0, stream>>>(g4, c1, lstm1A + HD, 2 * HD,
                                               dec_o + (size_t)t * HD, TT * HD);
  }

  // --- Final vocab projection: (2048 x 32000) ---
  gemm(stream, dec_o, HD, w_fc, HD, out, nullptr, VV, fc_b, BB * TT, VV, HD, 0);
}
